// MultiScaleImageFilters_26603027432173
// MI455X (gfx1250) — matrix operand order repaired and checked by reading, not compiled, not run on hardware
//
#include <hip/hip_runtime.h>
#include <stdint.h>

typedef _Float16 half8  __attribute__((ext_vector_type(8)));
typedef _Float16 half16 __attribute__((ext_vector_type(16)));
typedef float    float8 __attribute__((ext_vector_type(8)));
typedef unsigned int uint4v __attribute__((ext_vector_type(4)));
typedef int int4v  __attribute__((ext_vector_type(4)));
typedef int int8v  __attribute__((ext_vector_type(8)));

#if __has_builtin(__builtin_amdgcn_tensor_load_to_lds) && __has_builtin(__builtin_amdgcn_s_wait_tensorcnt)
#define HAVE_TDM 1
#else
#define HAVE_TDM 0
#endif

// ======================= compile-time filter tables =======================
// 26 (filter, dx) cases -> per-lane A-fragment (Toeplitz of kernel column dx)
// for V_WMMA_F32_16X16X32_F16. A 16-bit A 16x32 layout: lane L: m = L&15,
// hi = L>>4; element e in [0,16): K = 8*hi + e + (e<8 ? 0 : 8).
// A[m][k] = w[k - m - off][dx] (off = 3 - pad), zero outside band.

static constexpr double PI_ = 3.14159265358979323846;

static constexpr double cexp_(double x) {            // exp(x), x in [-5, 0]
  double t = x * 0.0625;                             // x/16
  double s = 1.0, term = 1.0;
  for (int i = 1; i < 24; ++i) { term *= t / i; s += term; }
  for (int i = 0; i < 4; ++i) s *= s;                // ^16
  return s;
}

static constexpr unsigned short f2h_(double v) {     // f64 -> f16 bits (RTN-ish)
  if (v == 0.0) return 0;
  unsigned short s = 0;
  if (v < 0) { s = 0x8000; v = -v; }
  int e = 0;
  while (v >= 2.0) { v *= 0.5; ++e; }
  while (v < 1.0)  { v *= 2.0; --e; }
  if (e > 15) return (unsigned short)(s | 0x7C00);
  if (e < -14) {
    int sh = -14 - e;
    double m = v;
    for (int i = 0; i < sh; ++i) m *= 0.5;
    int mant = (int)(m * 1024.0 + 0.5);
    if (mant >= 1024) return (unsigned short)(s | 0x0400);
    return (unsigned short)(s | mant);
  }
  int mant = (int)((v - 1.0) * 1024.0 + 0.5);
  int eb = e + 15;
  if (mant >= 1024) { mant = 0; ++eb; if (eb >= 31) return (unsigned short)(s | 0x7C00); }
  return (unsigned short)(s | (eb << 10) | mant);
}

struct ATab { unsigned short a[26][32][24]; };       // 24 shorts = 48B lane pitch

static constexpr ATab build_atab() {
  ATab t{};
  double w[6][7][7]{};
  // f0: lpf 3x3
  {
    const double l[3][3] = {{1,2,1},{2,4,2},{1,2,1}};
    for (int i = 0; i < 3; ++i) for (int j = 0; j < 3; ++j) w[0][i][j] = l[i][j] / 16.0;
  }
  // f1..f3: LoG 3/5/7
  {
    const int    ks[3] = {3, 5, 7};
    const double sg[3] = {0.8, 1.0, 1.4};
    for (int fi = 0; fi < 3; ++fi) {
      const int k = ks[fi];
      const double sig = sg[fi];
      const int h = k / 2;
      double sum = 0.0;
      for (int i = 0; i < k; ++i) for (int j = 0; j < k; ++j) {
        const double x = (double)(i - h), y = (double)(j - h);
        const double r2 = x * x + y * y;
        const double g   = cexp_(-r2 / (2.0 * sig * sig));
        const double lap = -1.0 / (PI_ * sig * sig * sig * sig) * (1.0 - r2 / (2.0 * sig * sig));
        const double v = lap * g;
        w[1 + fi][i][j] = v;
        sum += (v < 0 ? -v : v);
      }
      for (int i = 0; i < k; ++i) for (int j = 0; j < k; ++j) w[1 + fi][i][j] /= sum;
    }
  }
  // f4: hpf3, f5: hpf5
  for (int i = 0; i < 3; ++i) for (int j = 0; j < 3; ++j) w[4][i][j] = (i == 1 && j == 1) ? 8.0 : -1.0;
  for (int i = 0; i < 5; ++i) for (int j = 0; j < 5; ++j) w[5][i][j] = (i == 2 && j == 2) ? 24.0 : -1.0;

  unsigned short wh[6][7][7]{};
  for (int f = 0; f < 6; ++f)
    for (int i = 0; i < 7; ++i)
      for (int j = 0; j < 7; ++j) wh[f][i][j] = f2h_(w[f][i][j]);

  const int Ks[6]  = {3, 3, 5, 7, 3, 5};
  const int off[6] = {2, 2, 1, 0, 2, 1};             // off = 3 - pad
  int ci = 0;
  for (int f = 0; f < 6; ++f) {
    for (int dx = 0; dx < Ks[f]; ++dx) {
      for (int L = 0; L < 32; ++L) {
        const int hi = L >> 4, m = L & 15;
        for (int e = 0; e < 16; ++e) {
          const int K = 8 * hi + e + (e < 8 ? 0 : 8);
          const int dy = K - m - off[f];
          t.a[ci][L][e] = (dy >= 0 && dy < Ks[f]) ? wh[f][dy][dx] : (unsigned short)0;
        }
      }
      ++ci;
    }
  }
  return t;
}

static __constant__ ATab g_atab = build_atab();

// ======================= device helpers =======================

__device__ __forceinline__ half16 ldv16(const _Float16* p) {
  const half8* q = (const half8*)p;                  // 16B-aligned LDS
  half8 lo = q[0], hi = q[1];
  return __builtin_shufflevector(lo, hi, 0,1,2,3,4,5,6,7,8,9,10,11,12,13,14,15);
}

template <int KX, int OFF>
__device__ __forceinline__ float8 conv_tile(int cbase, const _Float16* afragLane,
                                            const half16 (&Bf)[7]) {
  float8 acc = {};
#pragma unroll
  for (int dx = 0; dx < KX; ++dx) {
    half16 A = ldv16(afragLane + (cbase + dx) * 768); // 768 shorts = case stride
    acc = __builtin_amdgcn_wmma_f32_16x16x32_f16(false, A, false, Bf[OFF + dx],
                                                 (short)0, acc, false, false);
  }
  return acc;
}

__device__ __forceinline__ float8 squash8(float8 r) {
  float8 o;
#pragma unroll
  for (int i = 0; i < 8; ++i) {
    float e = __expf(-0.2f * r[i]);                  // (tanh(.1r)+1)/2 == sigmoid(.2r)
    o[i] = 1.0f / (1.0f + e);
  }
  return o;
}

__device__ __forceinline__ float8 max8(float8 a, float8 b) {
  float8 o;
#pragma unroll
  for (int i = 0; i < 8; ++i) o[i] = fmaxf(a[i], b[i]);
  return o;
}

__device__ __forceinline__ void store8(float* p, float8 v) {
#pragma unroll
  for (int i = 0; i < 8; ++i) p[(size_t)i * 512] = v[i];
}

#if HAVE_TDM
__device__ __forceinline__ void tdm_load_tile(const float* gsrc, unsigned lds_off) {
  const unsigned long long ga = (unsigned long long)gsrc;
  uint4v g0;
  g0.x = 1u;                                         // count = 1 (valid descriptor)
  g0.y = lds_off;                                    // LDS byte address
  g0.z = (unsigned)ga;                               // global_addr[31:0]
  g0.w = (unsigned)(ga >> 32) | 0x80000000u;         // global_addr[56:32] | type=2
  int8v g1;
  g1[0] = (int)(2u << 16);                           // data_size = 2 (4 bytes)
  g1[1] = (int)(72u << 16);                          // tensor_dim0 = 72
  g1[2] = (int)(22u << 16);                          // tensor_dim1 = 22
  g1[3] = (int)(72u << 16);                          // tile_dim0 = 72
  g1[4] = 22;                                        // tile_dim1 = 22, tile_dim2 = 0
  g1[5] = 512;                                       // tensor_dim0_stride = 512 elems
  g1[6] = 0;
  g1[7] = 0;
  int4v gz4 = {0, 0, 0, 0};                          // groups 2/3 unused (2D tensor)
  int8v gz8 = {0, 0, 0, 0, 0, 0, 0, 0};              // extra group (clang-23 6-arg form)
  __builtin_amdgcn_tensor_load_to_lds(g0, g1, gz4, gz4, gz8, 0);
}
#endif

// ======================= kernels =======================

__global__ void copy_x_kernel(const float4* __restrict__ in, float4* __restrict__ out, int n4) {
  int i = blockIdx.x * blockDim.x + threadIdx.x;
  const int stride = gridDim.x * blockDim.x;
  for (; i < n4; i += stride) out[i] = in[i];
}

__global__ __launch_bounds__(128)
void MultiScaleImageFilters_26603027432173_kernel(const float* __restrict__ xin,
                                                  float* __restrict__ out_lpf,
                                                  float* __restrict__ out_log,
                                                  float* __restrict__ out_hpf) {
  __shared__ __align__(16) _Float16 afrag[26 * 768];   // 39936 B: A-fragments
  __shared__ __align__(16) _Float16 fbuf[70 * 40];     //  5600 B: col-major f16 input
  __shared__ __align__(16) float    stage[2][22 * 72]; // 12672 B: TDM f32 staging

  const int tid  = threadIdx.x;
  const int wave = tid >> 5;
  const int L    = tid & 31;
  const int L2   = L & 15;
  const int hi   = (L >> 4) & 1;

  const int plane = blockIdx.y;                 // 0..95
  const int SX    = blockIdx.x * 64;            // stripe base column
  const size_t pbase = (size_t)plane * 262144;  // 512*512

  const bool xint = (SX >= 3) && (SX + 68 <= 511);   // full stripe halo in-bounds

  // ---- once per workgroup: A-fragments const->LDS, zero f16 buffer ----
  {
    const uint4* src = (const uint4*)&g_atab;
    uint4*       dst = (uint4*)afrag;
    for (int i = tid; i < 2496; i += 128) dst[i] = src[i];
    for (int i = tid; i < 2800; i += 128) fbuf[i] = (_Float16)0.0f;
  }
  __syncthreads();

  const _Float16* aL = afrag + L * 24;

#if HAVE_TDM
  const unsigned st_off[2] = {
      (unsigned)(uintptr_t)(void*)&stage[0][0],
      (unsigned)(uintptr_t)(void*)&stage[1][0]};
#endif

  for (int it = 0; it < 32; ++it) {
    const int Y0 = it * 16;

#if HAVE_TDM
    if (wave == 0) __builtin_amdgcn_s_wait_tensorcnt(0);
#endif
    __syncthreads();   // staging ready AND previous tile's compute done

    // ---- stage input tile into col-major f16 buffer (cols 0..69, k 0..21) ----
    const bool tdm_tile = (HAVE_TDM != 0) && xint && (it >= 1) && (it <= 30);
    if (tdm_tile) {
#if HAVE_TDM
      const float* st = stage[it & 1];
      for (int e = tid; e < 1540; e += 128) {
        const int r = e / 70, c = e - r * 70;
        fbuf[c * 40 + r] = (_Float16)st[r * 72 + c];
      }
#endif
    } else {
      for (int e = tid; e < 1540; e += 128) {
        const int r = e / 70, c = e - r * 70;
        const int gy = Y0 - 3 + r, gx = SX - 3 + c;
        float v = 0.0f;
        if ((unsigned)gy < 512u && (unsigned)gx < 512u)
          v = xin[pbase + (size_t)gy * 512 + gx];
        fbuf[c * 40 + r] = (_Float16)v;
      }
    }
    __syncthreads();   // fbuf ready for all waves

#if HAVE_TDM
    // ---- prefetch next tile via Tensor Data Mover (overlaps with compute) ----
    if (wave == 0 && xint && (it + 1) <= 30) {
      const int j = it + 1;
      tdm_load_tile(xin + pbase + (size_t)(j * 16 - 3) * 512 + (SX - 3), st_off[j & 1]);
    }
#endif

    // ---- compute: 7 shared B fragments, 26 WMMAs across 6 filters ----
    half16 Bf[7];
#pragma unroll
    for (int s = 0; s < 7; ++s)
      Bf[s] = __builtin_shufflevector(*(const half8*)(fbuf + (wave * 16 + L2 + s) * 40 + 8 * hi), *(const half8*)(fbuf + (wave * 16 + L2 + s) * 40 + 16 + 8 * hi), 0,1,2,3,4,5,6,7,8,9,10,11,12,13,14,15);

    const size_t obase = pbase + (size_t)(Y0 + 8 * hi) * 512 + (size_t)(SX + wave * 16 + L2);

    float8 acc;
    acc = conv_tile<3, 2>(0, aL, Bf);                    // lpf (no squash)
    store8(out_lpf + obase, acc);

    acc = conv_tile<3, 2>(3, aL, Bf);                    // log3
    float8 lm = squash8(acc);
    acc = conv_tile<5, 1>(6, aL, Bf);                    // log5
    lm = max8(lm, squash8(acc));
    acc = conv_tile<7, 0>(11, aL, Bf);                   // log7
    lm = max8(lm, squash8(acc));
    store8(out_log + obase, lm);

    acc = conv_tile<3, 2>(18, aL, Bf);                   // hpf3
    float8 hm = squash8(acc);
    acc = conv_tile<5, 1>(21, aL, Bf);                   // hpf5
    hm = max8(hm, squash8(acc));
    store8(out_hpf + obase, hm);
  }
}

// ======================= launch =======================

extern "C" void kernel_launch(void* const* d_in, const int* in_sizes, int n_in,
                              void* d_out, int out_size, void* d_ws, size_t ws_size,
                              hipStream_t stream) {
  (void)in_sizes; (void)n_in; (void)out_size; (void)d_ws; (void)ws_size;

  const float* x   = (const float*)d_in[0];
  float*       out = (float*)d_out;
  const size_t N = 32ull * 3 * 512 * 512;   // 25165824 per output tensor

  // out[0] = x (identity copy)
  copy_x_kernel<<<2048, 256, 0, stream>>>((const float4*)x, (float4*)out, (int)(N / 4));

  // out[1]=lpf, out[2]=log, out[3]=hpf
  dim3 grid(8, 96);   // 8 x-stripes of 64 cols, 96 (n,c) planes
  MultiScaleImageFilters_26603027432173_kernel<<<grid, 128, 0, stream>>>(
      x, out + N, out + 2 * N, out + 3 * N);
}
